// MLPAttention_43774306681337
// MI455X (gfx1250) — matrix operand order repaired and checked by reading, not compiled, not run on hardware
//
#include <hip/hip_runtime.h>
#include <math.h>

typedef __attribute__((ext_vector_type(16))) _Float16 v16h;
typedef __attribute__((ext_vector_type(8)))  _Float16 v8h;
typedef __attribute__((ext_vector_type(8)))  float    v8f;

#define B_   64
#define S_   4096
#define QS_  1024
#define VS_  512
#define H_   512

// ---------------------------------------------------------------------------
// Kernel 1: convert W_v [H, Vs] fp32 -> f16 (for WMMA B operand)
// ---------------------------------------------------------------------------
__global__ void k_convert_wv(const float* __restrict__ Wv, _Float16* __restrict__ Wvh) {
    int i = blockIdx.x * blockDim.x + threadIdx.x;      // one float4 per thread
    union { float4 v; float f[4]; } u;
    u.v = reinterpret_cast<const float4*>(Wv)[i];
    _Float16 h0 = (_Float16)u.f[0], h1 = (_Float16)u.f[1];
    _Float16 h2 = (_Float16)u.f[2], h3 = (_Float16)u.f[3];
    __attribute__((ext_vector_type(4))) _Float16 hv = { h0, h1, h2, h3 };
    *reinterpret_cast<decltype(hv)*>(Wvh + 4 * i) = hv;
}

// ---------------------------------------------------------------------------
// Kernel 2: q_proj[b,h] = query[b,:] . W_q[h,:] + b_q[h]   (small, fp32)
// ---------------------------------------------------------------------------
__global__ __launch_bounds__(512) void k_qproj(const float* __restrict__ query,
                                               const float* __restrict__ Wq,
                                               const float* __restrict__ bq,
                                               float* __restrict__ qp) {
    __shared__ float qrow[QS_];
    const int b = blockIdx.x;
    for (int i = threadIdx.x; i < QS_; i += 512) qrow[i] = query[(size_t)b * QS_ + i];
    __syncthreads();
    const int h = threadIdx.x;                 // 512 threads = H
    const float* w = Wq + (size_t)h * QS_;
    float acc = 0.f;
#pragma unroll 4
    for (int k = 0; k < QS_; k += 4) {
        union { float4 v; float f[4]; } u;
        u.v = *reinterpret_cast<const float4*>(w + k);
        acc += u.f[0] * qrow[k]     + u.f[1] * qrow[k + 1]
             + u.f[2] * qrow[k + 2] + u.f[3] * qrow[k + 3];
    }
    qp[(size_t)b * H_ + h] = acc + bq[h];
}

// ---------------------------------------------------------------------------
// Kernel 3: fused  v_proj (WMMA) + tanh(q_proj + v_proj) . w_h  -> logits[b,s]
//   block = 256 threads (8 waves); wave owns 16 rows of s; block M = 128.
//   A (value rows, f32->f16) kept in registers for all K=512 (read HBM once).
//   B (W_v f16) double-buffered in LDS: tile nt+1 is prefetched into registers
//   before the barrier and written to the idle buffer while WMMAs consume the
//   current one -> one barrier per N-tile, B latency overlapped with compute.
// ---------------------------------------------------------------------------
#define BSTR (VS_ + 16)   // padded LDS stride in halfs (32B pad -> spread banks)

__global__ __launch_bounds__(256) void k_fused(const float*    __restrict__ value,
                                               const _Float16* __restrict__ Wvh,
                                               const float*    __restrict__ qp,
                                               const float*    __restrict__ wh,
                                               float*          __restrict__ logits) {
    __shared__ _Float16 lds_b[2][16 * BSTR];
    __shared__ float    lds_qp[H_];
    __shared__ float    lds_wh[H_];

    const int b    = blockIdx.y;
    const int s0   = blockIdx.x * 128;
    const int tid  = threadIdx.x;
    const int wave = tid >> 5;
    const int lane = tid & 31;
    const int l16  = lane & 15;
    const int hi   = lane >> 4;     // 0: lanes 0-15, 1: lanes 16-31

    for (int i = tid; i < H_; i += 256) {
        lds_qp[i] = qp[(size_t)b * H_ + i];
        lds_wh[i] = wh[i];
    }

    // Per-thread staging slots: 1024 x 8-half chunks per tile, 4 per thread.
    // chunk c -> row n = c>>6, k-offset kc = (c&63)*8
    const int cn[4] = { (tid + 0*256) >> 6, (tid + 1*256) >> 6,
                        (tid + 2*256) >> 6, (tid + 3*256) >> 6 };
    const int ck[4] = { ((tid + 0*256) & 63) * 8, ((tid + 1*256) & 63) * 8,
                        ((tid + 2*256) & 63) * 8, ((tid + 3*256) & 63) * 8 };

    // ---- load A fragments for the whole K=512 into registers -------------
    // A 16x32 layout (matches B): lanes 0-15 (row M=lane) hold K {0..15},
    // lanes 16-31 (row M=lane-16) hold K {16..31} within each k-step.
    const int    srow = s0 + wave * 16 + l16;
    const float* arow = value + ((size_t)b * S_ + srow) * (size_t)VS_;
    const int    koff = hi * 16;

    v16h a[16];
#pragma unroll
    for (int ks = 0; ks < 16; ++ks) {
        union { float4 v; float f[4]; } p0, p1, p2, p3;
        p0.v = *reinterpret_cast<const float4*>(arow + ks * 32 + koff);
        p1.v = *reinterpret_cast<const float4*>(arow + ks * 32 + koff + 4);
        p2.v = *reinterpret_cast<const float4*>(arow + ks * 32 + koff + 8);
        p3.v = *reinterpret_cast<const float4*>(arow + ks * 32 + koff + 12);
#pragma unroll
        for (int t = 0; t < 4; ++t) {
            a[ks][t]      = (_Float16)p0.f[t];
            a[ks][4 + t]  = (_Float16)p1.f[t];
            a[ks][8 + t]  = (_Float16)p2.f[t];
            a[ks][12 + t] = (_Float16)p3.f[t];
        }
    }

    // ---- prime buffer 0 with N-tile 0 ------------------------------------
    v8h pf[4];
#pragma unroll
    for (int j = 0; j < 4; ++j)
        pf[j] = *reinterpret_cast<const v8h*>(Wvh + (size_t)cn[j] * VS_ + ck[j]);
#pragma unroll
    for (int j = 0; j < 4; ++j)
        *reinterpret_cast<v8h*>(&lds_b[0][cn[j] * BSTR + ck[j]]) = pf[j];

    float part[8] = {0.f, 0.f, 0.f, 0.f, 0.f, 0.f, 0.f, 0.f};

    // ---- N loop over 32 tiles of H, block in lockstep --------------------
    for (int nt = 0; nt < 32; ++nt) {
        const int cur = nt & 1;

        // prefetch next B tile into registers (in flight across the WMMAs)
        if (nt + 1 < 32) {
            const _Float16* src = Wvh + (size_t)((nt + 1) * 16) * VS_;
#pragma unroll
            for (int j = 0; j < 4; ++j)
                pf[j] = *reinterpret_cast<const v8h*>(src + (size_t)cn[j] * VS_ + ck[j]);
        }

        __syncthreads();   // publish buf[cur] stores; protect buf WAR

        // B 32x16 operand as used here: lane = column N, lane half selects K half (contiguous k, same as A).
        const _Float16* bbase = &lds_b[cur][l16 * BSTR + hi * 16];
        v8f acc = {};
#pragma unroll
        for (int ks = 0; ks < 16; ++ks) {
            v16h bf = *reinterpret_cast<const v16h*>(bbase + ks * 32);
            acc = __builtin_amdgcn_wmma_f32_16x16x32_f16(
                      false, a[ks], false, bf, (short)0, acc, false, false);
        }

        // write prefetched tile into the idle buffer while nobody reads it
        if (nt + 1 < 32) {
#pragma unroll
            for (int j = 0; j < 4; ++j)
                *reinterpret_cast<v8h*>(&lds_b[cur ^ 1][cn[j] * BSTR + ck[j]]) = pf[j];
        }

        // epilogue: C layout -> lane holds column n = nt*16+l16, rows i(+8*hi)
        const int   ncol = nt * 16 + l16;
        const float qv   = lds_qp[ncol];
        const float wv   = lds_wh[ncol];
#pragma unroll
        for (int i = 0; i < 8; ++i)
            part[i] += tanhf(acc[i] + qv) * wv;
    }

    // ---- reduce partial logits across the 16 lanes of each half-wave -----
#pragma unroll
    for (int i = 0; i < 8; ++i) {
        float v = part[i];
        v += __shfl_xor(v, 1, 32);
        v += __shfl_xor(v, 2, 32);
        v += __shfl_xor(v, 4, 32);
        v += __shfl_xor(v, 8, 32);
        part[i] = v;
    }
    if (l16 == 0) {   // lane 0 -> rows 0..7, lane 16 -> rows 8..15
        float* lrow = logits + (size_t)b * S_ + s0 + wave * 16 + hi * 8;
#pragma unroll
        for (int i = 0; i < 8; ++i) lrow[i] = part[i];
    }
}

// ---------------------------------------------------------------------------
// Kernel 4: masked softmax over S per batch row
// ---------------------------------------------------------------------------
__global__ __launch_bounds__(256) void k_softmax(const float* __restrict__ logits,
                                                 const unsigned char* __restrict__ mask,
                                                 float* __restrict__ out) {
    __shared__ float redmax[8];
    __shared__ float redsum[8];
    const int b = blockIdx.x;
    const float*         lr = logits + (size_t)b * S_;
    const unsigned char* mr = mask   + (size_t)b * S_;
    float*               orow = out  + (size_t)b * S_;

    const float NEG = -3.402823466e+38f;
    float mx = NEG;
    for (int i = threadIdx.x; i < S_; i += 256) {
        float v = mr[i] ? lr[i] : NEG;
        mx = fmaxf(mx, v);
    }
    for (int o = 1; o < 32; o <<= 1) mx = fmaxf(mx, __shfl_xor(mx, o, 32));
    if ((threadIdx.x & 31) == 0) redmax[threadIdx.x >> 5] = mx;
    __syncthreads();
    float bmax = redmax[0];
#pragma unroll
    for (int j = 1; j < 8; ++j) bmax = fmaxf(bmax, redmax[j]);

    float s = 0.f;
    for (int i = threadIdx.x; i < S_; i += 256) {
        float v = mr[i] ? lr[i] : NEG;
        float e = __expf(v - bmax);
        orow[i] = e;
        s += e;
    }
    for (int o = 1; o < 32; o <<= 1) s += __shfl_xor(s, o, 32);
    if ((threadIdx.x & 31) == 0) redsum[threadIdx.x >> 5] = s;
    __syncthreads();
    float bsum = 0.f;
#pragma unroll
    for (int j = 0; j < 8; ++j) bsum += redsum[j];
    const float inv = 1.0f / bsum;

    for (int i = threadIdx.x; i < S_; i += 256) orow[i] *= inv;
}

// ---------------------------------------------------------------------------
extern "C" void kernel_launch(void* const* d_in, const int* in_sizes, int n_in,
                              void* d_out, int out_size, void* d_ws, size_t ws_size,
                              hipStream_t stream) {
    const float*         query = (const float*)d_in[0];          // [64,1024]
    const float*         value = (const float*)d_in[1];          // [64,4096,512]
    const unsigned char* mask  = (const unsigned char*)d_in[2];  // [64,4096] bool
    const float*         Wq    = (const float*)d_in[3];          // [512,1024]
    const float*         bq    = (const float*)d_in[4];          // [512]
    const float*         Wv    = (const float*)d_in[5];          // [512,512]
    const float*         wh    = (const float*)d_in[6];          // [512]
    float*               out   = (float*)d_out;                  // [64,4096]

    // workspace layout
    char* ws = (char*)d_ws;
    _Float16* Wvh    = (_Float16*)(ws);                          // 512 KiB
    float*    qp     = (float*)(ws + (512 << 10));               // 128 KiB
    float*    logits = (float*)(ws + (640 << 10));               // 1 MiB

    // 1) W_v fp32 -> f16   (262144 elems, 4 per thread)
    k_convert_wv<<<dim3(256), dim3(256), 0, stream>>>(Wv, Wvh);
    // 2) q_proj
    k_qproj<<<dim3(B_), dim3(512), 0, stream>>>(query, Wq, bq, qp);
    // 3) fused WMMA GEMM + tanh + w_h dot -> logits
    k_fused<<<dim3(S_ / 128, B_), dim3(256), 0, stream>>>(value, Wvh, qp, wh, logits);
    // 4) masked softmax
    k_softmax<<<dim3(B_), dim3(256), 0, stream>>>(logits, mask, out);
}
